// ALGA_33277406609499
// MI455X (gfx1250) — matrix operand order repaired and checked by reading, not compiled, not run on hardware
//
#include <hip/hip_runtime.h>
#include <math.h>

typedef __attribute__((ext_vector_type(16))) _Float16 v16h;
typedef __attribute__((ext_vector_type(8)))  _Float16 h8;
typedef __attribute__((ext_vector_type(4)))  _Float16 h4;
typedef __attribute__((ext_vector_type(8)))  float    v8f;
typedef __attribute__((ext_vector_type(4)))  float    f4;

#define N_PIX 4096
#define BATCH 4
#define HEADS 3
#define BH (BATCH * HEADS)
#define LOG2E 1.4426950408889634f

// ---------------- Kernel 1: per-batch luma stats (mu, std of |L-mu| ddof=1) ----
__global__ void ALGA_luma_stats(const float* __restrict__ rgb,
                                float* __restrict__ mu_out,
                                float* __restrict__ std_out) {
  const int b = blockIdx.x;
  const int tid = threadIdx.x;
  const float* r = rgb + (size_t)b * 3 * N_PIX;
  __shared__ float red[256];
  __shared__ float mu_s, s1_s;

  float s = 0.f;
  for (int i = tid; i < N_PIX; i += 256) {
    float L = 0.299f * r[i] + 0.587f * r[N_PIX + i] + 0.114f * r[2 * N_PIX + i];
    s += L;
  }
  red[tid] = s; __syncthreads();
  for (int off = 128; off > 0; off >>= 1) {
    if (tid < off) red[tid] += red[tid + off];
    __syncthreads();
  }
  if (tid == 0) mu_s = red[0] / (float)N_PIX;
  __syncthreads();
  const float mu = mu_s;

  float s1 = 0.f, s2 = 0.f;
  for (int i = tid; i < N_PIX; i += 256) {
    float L = 0.299f * r[i] + 0.587f * r[N_PIX + i] + 0.114f * r[2 * N_PIX + i];
    float d = fabsf(L - mu);
    s1 += d; s2 += d * d;
  }
  red[tid] = s1; __syncthreads();
  for (int off = 128; off > 0; off >>= 1) {
    if (tid < off) red[tid] += red[tid + off];
    __syncthreads();
  }
  if (tid == 0) s1_s = red[0];
  __syncthreads();
  red[tid] = s2; __syncthreads();
  for (int off = 128; off > 0; off >>= 1) {
    if (tid < off) red[tid] += red[tid + off];
    __syncthreads();
  }
  if (tid == 0) {
    float mean_d = s1_s / (float)N_PIX;
    float var = (red[0] - (float)N_PIX * mean_d * mean_d) / (float)(N_PIX - 1);
    if (var < 0.f) var = 0.f;
    mu_out[b] = mu;
    std_out[b] = sqrtf(var) + 1e-6f;
  }
}

// ---------------- Kernel 2: q/k/v 1x1 convs + luma gate on q --------------------
__global__ void ALGA_qkv(const float* __restrict__ rgb,
                         const float* __restrict__ wq,
                         const float* __restrict__ wk,
                         const float* __restrict__ wv,
                         const float* __restrict__ mu,
                         const float* __restrict__ stdv,
                         float* __restrict__ q, float* __restrict__ k,
                         float* __restrict__ v) {
  int idx = blockIdx.x * blockDim.x + threadIdx.x;
  if (idx >= BATCH * N_PIX) return;
  const int b = idx / N_PIX, n = idx - b * N_PIX;
  const float* r = rgb + (size_t)b * 3 * N_PIX;
  const float r0 = r[n], r1 = r[N_PIX + n], r2 = r[2 * N_PIX + n];
  const float L = 0.299f * r0 + 0.587f * r1 + 0.114f * r2;
  const float d = fabsf(L - mu[b]);
  const float gate = 1.f / (1.f + __expf(-d / stdv[b]));
  const float qs = 1.f + gate;
#pragma unroll
  for (int o = 0; o < 3; ++o) {
    const size_t oi = (size_t)(b * 3 + o) * N_PIX + n;
    q[oi] = (wq[o * 3 + 0] * r0 + wq[o * 3 + 1] * r1 + wq[o * 3 + 2] * r2) * qs;
    k[oi] = wk[o * 3 + 0] * r0 + wk[o * 3 + 1] * r1 + wk[o * 3 + 2] * r2;
    v[oi] = wv[o * 3 + 0] * r0 + wv[o * 3 + 1] * r1 + wv[o * 3 + 2] * r2;
  }
}

// ---------------- Kernel 3: per-(b,h) k min/max for analytic row-max ------------
__global__ void ALGA_kminmax(const float* __restrict__ kg,
                             float* __restrict__ kmax, float* __restrict__ kmin) {
  const int bh = blockIdx.x;
  const int tid = threadIdx.x;
  __shared__ float rmax[256], rmin[256];
  const size_t base = (size_t)bh * N_PIX;
  float mx = -1e30f, mn = 1e30f;
  for (int i = tid; i < N_PIX; i += 256) {
    float x = kg[base + i];
    mx = fmaxf(mx, x);
    mn = fminf(mn, x);
  }
  rmax[tid] = mx; rmin[tid] = mn; __syncthreads();
  for (int off = 128; off > 0; off >>= 1) {
    if (tid < off) {
      rmax[tid] = fmaxf(rmax[tid], rmax[tid + off]);
      rmin[tid] = fminf(rmin[tid], rmin[tid + off]);
    }
    __syncthreads();
  }
  if (tid == 0) { kmax[bh] = rmax[0]; kmin[bh] = rmin[0]; }
}

// ---------------- Kernel 4: rank-1 softmax attention via WMMA -------------------
// A (16x32 f16) = [v; 1; 0...]: row 0 = v chunk, row 1 = ones.
// B (32x16 f16) = exp(q_n * k_m - M_n), column n = lane&15, K half = lane>>4.
// D = A*B accumulated in f32: D[0][n] = numerator, D[1][n] = denominator,
// landing in C VGPR0/VGPR1 on lanes 0-15 (N = lane) -> result = c0 * rcp(c1).
// A-tile is fetched through per-lane pointers: row 0 walks v_lds, row 1 reads a
// constant ones-block, rows>=2 read a zeros-block -> no cndmask in the loop.
__global__ void ALGA_attn(const float* __restrict__ qg,
                          const float* __restrict__ kg,
                          const float* __restrict__ vg,
                          const float* __restrict__ kmax,
                          const float* __restrict__ kmin,
                          float* __restrict__ ao) {
  __shared__ float    k_lds[N_PIX];   // 16 KB
  __shared__ _Float16 v_lds[N_PIX];   // 8 KB
  __shared__ _Float16 czero[8];
  __shared__ _Float16 cone[8];
  const int bh  = blockIdx.y;
  const int tid = threadIdx.x;        // 0..127 (4 waves)
  const size_t base = (size_t)bh * N_PIX;

  if (tid < 8) { czero[tid] = (_Float16)0.f; cone[tid] = (_Float16)1.f; }
  // Vectorized staging: b128 global loads -> b128/b64 LDS stores.
  {
    const f4* kgv = (const f4*)(kg + base);
    const f4* vgv = (const f4*)(vg + base);
    f4* klv = (f4*)k_lds;
    h4* vlv = (h4*)v_lds;
    for (int i = tid; i < N_PIX / 4; i += 128) {
      klv[i] = kgv[i];
      const f4 vv = vgv[i];
      h4 hv;
#pragma unroll
      for (int t = 0; t < 4; ++t) hv[t] = (_Float16)vv[t];
      vlv[i] = hv;
    }
  }
  __syncthreads();

  const int wave = tid >> 5;
  const int lane = tid & 31;
  const int g    = lane >> 4;       // K half
  const int row  = lane & 15;       // A row M / B column N
  const int n0   = blockIdx.x * 64 + wave * 16;

  const float qr   = qg[base + n0 + row];
  const float kmx  = kmax[bh], kmn = kmin[bh];
  const float Mrow = (qr >= 0.f) ? qr * kmx : qr * kmn;
  // exp(q*k - M) = exp2(q2*k - M2): fold log2(e) into the FMA operands.
  const float q2 = qr * LOG2E;
  const float M2 = Mrow * LOG2E;

  // Per-lane A-tile pointers (uniform within a lane, no selects in the loop).
  const _Float16* ap0;
  const _Float16* ap1;
  int astep;
  if (row == 0)      { ap0 = &v_lds[16 * g]; ap1 = &v_lds[16 * g + 8]; astep = 32; }
  else if (row == 1) { ap0 = cone;  ap1 = cone;  astep = 0; }
  else               { ap0 = czero; ap1 = czero; astep = 0; }

  v8f c = {};
  for (int m0 = 0; m0 < N_PIX; m0 += 32) {
    // ---- B tile (exp): element e <-> K = 16*g + e (contiguous, b128 loads)
    const f4* kp = (const f4*)&k_lds[m0 + 16 * g];
    float arg[16];
#pragma unroll
    for (int j = 0; j < 4; ++j) {
      const f4 kk = kp[j];
#pragma unroll
      for (int t = 0; t < 4; ++t)
        arg[4 * j + t] = __builtin_fmaf(q2, kk[t], -M2);
    }
    v16h bm;
#pragma unroll
    for (int e = 0; e < 16; ++e)
      bm[e] = (_Float16)__builtin_amdgcn_exp2f(arg[e]);

    // ---- A tile ([v;1;0...]) via per-lane pointers, two b128 LDS loads
    const h8 va0 = *(const h8*)ap0;
    const h8 va1 = *(const h8*)ap1;
    ap0 += astep; ap1 += astep;
    v16h a;
#pragma unroll
    for (int e = 0; e < 8; ++e) { a[e] = va0[e]; a[e + 8] = va1[e]; }

    c = __builtin_amdgcn_wmma_f32_16x16x32_f16(
        /*neg_a=*/false, a, /*neg_b=*/false, bm,
        /*c_mod=*/(short)0, c, /*reuse_a=*/false, /*reuse_b=*/false);
  }

  // Denominator >= 1 (max-subtracted softmax), so fast reciprocal is safe.
  if (lane < 16) ao[base + n0 + lane] = c[0] * __builtin_amdgcn_rcpf(c[1]);
}

// ---------------- Kernel 5: wo mix + residual + clip ---------------------------
__global__ void ALGA_out(const float* __restrict__ rgb,
                         const float* __restrict__ wo,
                         const float* __restrict__ ao,
                         float* __restrict__ out) {
  int idx = blockIdx.x * blockDim.x + threadIdx.x;
  if (idx >= BATCH * N_PIX) return;
  const int b = idx / N_PIX, n = idx - b * N_PIX;
  const float a0 = ao[(size_t)(b * 3 + 0) * N_PIX + n];
  const float a1 = ao[(size_t)(b * 3 + 1) * N_PIX + n];
  const float a2 = ao[(size_t)(b * 3 + 2) * N_PIX + n];
#pragma unroll
  for (int o = 0; o < 3; ++o) {
    const float y = wo[o * 3 + 0] * a0 + wo[o * 3 + 1] * a1 + wo[o * 3 + 2] * a2;
    const size_t oi = (size_t)(b * 3 + o) * N_PIX + n;
    float val = rgb[oi] + y;
    out[oi] = fminf(fmaxf(val, 0.f), 1.f);
  }
}

extern "C" void kernel_launch(void* const* d_in, const int* in_sizes, int n_in,
                              void* d_out, int out_size, void* d_ws, size_t ws_size,
                              hipStream_t stream) {
  const float* rgb = (const float*)d_in[0];
  const float* wq  = (const float*)d_in[1];
  const float* wk  = (const float*)d_in[2];
  const float* wv  = (const float*)d_in[3];
  const float* wo  = (const float*)d_in[4];

  float* ws   = (float*)d_ws;
  float* mu   = ws;            // 4
  float* stdv = ws + 4;        // 4
  float* kmax = ws + 8;        // 12
  float* kmin = ws + 20;       // 12
  float* q    = ws + 32;                 // BH*N
  float* k    = q + (size_t)BH * N_PIX;  // BH*N
  float* v    = k + (size_t)BH * N_PIX;  // BH*N
  float* ao   = v + (size_t)BH * N_PIX;  // BH*N

  ALGA_luma_stats<<<BATCH, 256, 0, stream>>>(rgb, mu, stdv);
  ALGA_qkv<<<(BATCH * N_PIX + 255) / 256, 256, 0, stream>>>(rgb, wq, wk, wv, mu,
                                                            stdv, q, k, v);
  ALGA_kminmax<<<BH, 256, 0, stream>>>(k, kmax, kmin);
  ALGA_attn<<<dim3(N_PIX / 64, BH), 128, 0, stream>>>(q, k, v, kmax, kmin, ao);
  ALGA_out<<<(BATCH * N_PIX + 255) / 256, 256, 0, stream>>>(rgb, wo, ao,
                                                            (float*)d_out);
}
